// TopK_Pool_57930518889060
// MI455X (gfx1250) — compile-verified, matrix operand order checked by reading, not run on hardware
//
#include <hip/hip_runtime.h>
#include <hip/hip_bf16.h>
#include <math.h>

// Problem constants (match reference: B=64, N_PER=2048, C=512, K=512)
#define NUM_GRAPHS   64
#define NODES_PER_G  2048
#define CDIM         512
#define KSEL         512
#define NTOTAL       (NUM_GRAPHS * NODES_PER_G)    // 131072 nodes
#define NOUT_ROWS    (NUM_GRAPHS * KSEL)           // 32768 selected rows
#define OUT_ELEMS    (NOUT_ROWS * CDIM)            // 16777216 floats of `out`

typedef __attribute__((ext_vector_type(2)))  float    v2f;
typedef __attribute__((ext_vector_type(8)))  float    v8f;
typedef __attribute__((ext_vector_type(16))) _Float16 v16h;

// ---------------------------------------------------------------------------
// Kernel 0: Wn = W / ||W||   (512 elements, one block)
// ---------------------------------------------------------------------------
__global__ __launch_bounds__(512)
void normalize_w_kernel(const float* __restrict__ W, float* __restrict__ Wn) {
    __shared__ float red[512];
    const int t = threadIdx.x;
    const float w = W[t];
    red[t] = w * w;
    __syncthreads();
    for (int s = 256; s > 0; s >>= 1) {
        if (t < s) red[t] += red[t + s];
        __syncthreads();
    }
    const float inv = 1.0f / sqrtf(red[0]);
    Wn[t] = w * inv;
}

// ---------------------------------------------------------------------------
// Kernel 1: per-node scores via WMMA.
//   D = A x B + C with A = 16 nodes x 4 C-chunk (f32), B[k][n] = Wn[k] for all
//   n (replicated columns) -> every column of D is the 16 node scores.
//   8 waves / block, 16 nodes / wave -> 128 nodes / block, 1024 blocks.
// ---------------------------------------------------------------------------
#define WAVES_PER_BLOCK 8
__global__ __launch_bounds__(WAVES_PER_BLOCK * 32)
void score_kernel(const float* __restrict__ feat0, const float* __restrict__ Wn,
                  float* __restrict__ y) {
    const int lane = threadIdx.x & 31;
    const int wave = threadIdx.x >> 5;
    const int node_base = (blockIdx.x * WAVES_PER_BLOCK + wave) * 16;
    const int m     = lane & 15;          // M index within tile
    const int khalf = (lane >> 4) * 2;    // K sub-offset for this lane half

    v8f c = {0.f, 0.f, 0.f, 0.f, 0.f, 0.f, 0.f, 0.f};

#if __has_builtin(__builtin_amdgcn_wmma_f32_16x16x4_f32)
    // Full-f32 path: A frag = float2 {K=khalf, K=khalf+1} of this node's row,
    // B frag = same two elements of normalized W (column-replicated).
    const float* arow = feat0 + (size_t)(node_base + m) * CDIM + khalf;
    const float* bvec = Wn + khalf;
    #pragma unroll 8
    for (int s = 0; s < CDIM / 4; ++s) {
        v2f a = *(const v2f*)(arow + s * 4);
        v2f b = *(const v2f*)(bvec + s * 4);
        c = __builtin_amdgcn_wmma_f32_16x16x4_f32(false, a, false, b,
                                                  (short)0, c, false, false);
    }
#else
    // Fallback: f16 16x16x32 (convert on the fly).
    const int kbA = (lane >> 4) * 8;      // A: regs 0-3 -> K=kbA+0..7, regs 4-7 -> +16
    const int kbB = (lane >> 4) * 16;     // B: 16 halves -> K=kbB+0..15
    const float* arow16 = feat0 + (size_t)(node_base + m) * CDIM;
    for (int s = 0; s < CDIM / 32; ++s) {
        const int kk = s * 32;
        v16h a, b;
        #pragma unroll
        for (int t = 0; t < 8; ++t) {
            a[t]     = (_Float16)arow16[kk + kbA + t];
            a[8 + t] = (_Float16)arow16[kk + kbA + 16 + t];
        }
        #pragma unroll
        for (int t = 0; t < 16; ++t)
            b[t] = (_Float16)Wn[kk + kbB + t];
        c = __builtin_amdgcn_wmma_f32_16x16x32_f16(false, a, false, b,
                                                   (short)0, c, false, false);
    }
#endif

    // D layout: lane L, VGPR v  ->  M = v + 8*(L/16), N = L%16 (all N equal).
    if (m == 0) {
        const int mb = (lane >> 4) * 8;
        #pragma unroll
        for (int v = 0; v < 8; ++v)
            y[node_base + mb + v] = c[v];
    }
}

// ---------------------------------------------------------------------------
// Kernel 2: per-graph top-k by rank counting (stable, lower-index tie-break),
// emitted in ascending-index order via chunked prefix sum. One block / graph.
// ---------------------------------------------------------------------------
__global__ __launch_bounds__(1024)
void topk_kernel(const float* __restrict__ y, float* __restrict__ gates,
                 int* __restrict__ nidx, float* __restrict__ flat_idx_out) {
    __shared__ float ys[NODES_PER_G];
    __shared__ int   sel[NODES_PER_G];
    __shared__ int   csum[NODES_PER_G / 32];
    const int g   = blockIdx.x;
    const int tid = threadIdx.x;

    for (int i = tid; i < NODES_PER_G; i += 1024)
        ys[i] = y[g * NODES_PER_G + i];
    __syncthreads();

    // rank_i = #{j : y_j > y_i} + #{j < i : y_j == y_i}; keep if rank < k.
    #pragma unroll
    for (int e = 0; e < 2; ++e) {
        const int i = tid + e * 1024;
        const float yi = ys[i];
        int rank = 0;
        for (int j = 0; j < NODES_PER_G; ++j) {
            const float yj = ys[j];                 // uniform addr -> LDS broadcast
            rank += (int)((yj > yi) || ((yj == yi) && (j < i)));
        }
        sel[i] = (rank < KSEL) ? 1 : 0;
    }
    __syncthreads();

    if (tid < NODES_PER_G / 32) {
        int s = 0;
        for (int t = 0; t < 32; ++t) s += sel[tid * 32 + t];
        csum[tid] = s;
    }
    __syncthreads();

    #pragma unroll
    for (int e = 0; e < 2; ++e) {
        const int i = tid + e * 1024;
        if (sel[i]) {
            int pos = 0;
            for (int cc = 0; cc < (i >> 5); ++cc) pos += csum[cc];
            for (int j = (i & ~31); j < i; ++j)   pos += sel[j];
            const int r  = g * KSEL + pos;          // output row (index-sorted)
            const int fi = g * NODES_PER_G + i;     // flat node index
            const float yi = ys[i];
            gates[r]        = 1.0f / (1.0f + expf(-yi));   // sigmoid gate
            nidx[r]         = fi;
            flat_idx_out[r] = (float)fi;            // second tuple output
        }
    }
}

// ---------------------------------------------------------------------------
// Kernel 3: out[r, :] = gate[r] * feat0[nidx[r], :]
// One block per row; 128 threads x float4 = 512 floats, fully coalesced b128.
// ---------------------------------------------------------------------------
__global__ __launch_bounds__(128)
void gather_kernel(const float* __restrict__ feat0, const float* __restrict__ gates,
                   const int* __restrict__ nidx, float* __restrict__ out) {
    const int r     = blockIdx.x;
    const int node  = nidx[r];
    const float gsc = gates[r];
    const float4* src = (const float4*)(feat0 + (size_t)node * CDIM);
    float4*       dst = (float4*)(out + (size_t)r * CDIM);
    float4 v = src[threadIdx.x];
    v.x *= gsc; v.y *= gsc; v.z *= gsc; v.w *= gsc;
    dst[threadIdx.x] = v;
}

// ---------------------------------------------------------------------------
extern "C" void kernel_launch(void* const* d_in, const int* in_sizes, int n_in,
                              void* d_out, int out_size, void* d_ws, size_t ws_size,
                              hipStream_t stream) {
    (void)in_sizes; (void)n_in; (void)out_size; (void)ws_size;
    const float* feat0 = (const float*)d_in[0];   // [131072, 512] f32
    const float* W     = (const float*)d_in[1];   // [512] f32
    float* out = (float*)d_out;

    // Workspace layout (all 4B-aligned):
    char*  ws    = (char*)d_ws;
    float* Wn    = (float*)(ws);                                       // 512 f32
    float* yscr  = (float*)(ws + 2048);                                // 131072 f32
    float* gates = (float*)(ws + 2048 + (size_t)NTOTAL * 4);           // 32768 f32
    int*   nidx  = (int*)  (ws + 2048 + (size_t)NTOTAL * 4
                               + (size_t)NOUT_ROWS * 4);               // 32768 i32
    float* flat_out = out + OUT_ELEMS;  // flat_idx tuple output (as f32)

    hipLaunchKernelGGL(normalize_w_kernel, dim3(1), dim3(512), 0, stream, W, Wn);
    hipLaunchKernelGGL(score_kernel,
                       dim3(NTOTAL / (WAVES_PER_BLOCK * 16)),
                       dim3(WAVES_PER_BLOCK * 32), 0, stream, feat0, Wn, yscr);
    hipLaunchKernelGGL(topk_kernel, dim3(NUM_GRAPHS), dim3(1024), 0, stream,
                       yscr, gates, nidx, flat_out);
    hipLaunchKernelGGL(gather_kernel, dim3(NOUT_ROWS), dim3(128), 0, stream,
                       feat0, gates, nidx, out);
}
